// Attention_67817533604369
// MI455X (gfx1250) — matrix operand order repaired and checked by reading, not compiled, not run on hardware
//
#include <hip/hip_runtime.h>

typedef __attribute__((ext_vector_type(16))) _Float16 v16h;
typedef __attribute__((ext_vector_type(8)))  _Float16 v8h;
typedef __attribute__((ext_vector_type(8)))  float    v8f;

#define BATCH 2
#define TSEQ  2048
#define CDIM  512
#define NH    8
#define HD    64
#define INNER 512
#define NQKV  1536

static __device__ __forceinline__ v16h cat16(v8h lo, v8h hi) {
  return __builtin_shufflevector(lo, hi, 0,1,2,3,4,5,6,7,8,9,10,11,12,13,14,15);
}

static __device__ __forceinline__ v8f wmma_f16(v16h a, v16h b, v8f c) {
  return __builtin_amdgcn_wmma_f32_16x16x32_f16(false, a, false, b, (short)0, c, false, false);
}

// Async-copy 512B (16B per lane) from global to LDS. dst = LDS byte address.
static __device__ __forceinline__ void async_copy_b128(unsigned lds_dst,
                                                       const _Float16* gsrc) {
  asm volatile("global_load_async_to_lds_b128 %0, %1, off"
               :: "v"(lds_dst), "v"(gsrc) : "memory");
}
static __device__ __forceinline__ void wait_async0() {
  asm volatile("s_wait_asynccnt 0x0" ::: "memory");
}

// ---------------- LayerNorm (fp32 in, f16 out) ----------------
__global__ void ln_to_f16(const float* __restrict__ x, const float* __restrict__ g,
                          const float* __restrict__ be, _Float16* __restrict__ xn) {
  __shared__ float s1[256];
  __shared__ float s2[256];
  int row = blockIdx.x;
  int tid = threadIdx.x;
  const float* xr = x + (size_t)row * CDIM;
  float a = xr[tid];
  float b = xr[tid + 256];
  s1[tid] = a + b;
  s2[tid] = a * a + b * b;
  __syncthreads();
  for (int off = 128; off > 0; off >>= 1) {
    if (tid < off) { s1[tid] += s1[tid + off]; s2[tid] += s2[tid + off]; }
    __syncthreads();
  }
  float mu  = s1[0] * (1.0f / CDIM);
  float var = s2[0] * (1.0f / CDIM) - mu * mu;
  float rs  = rsqrtf(var + 1e-5f);
  _Float16* xo = xn + (size_t)row * CDIM;
  xo[tid]       = (_Float16)((a - mu) * rs * g[tid]       + be[tid]);
  xo[tid + 256] = (_Float16)((b - mu) * rs * g[tid + 256] + be[tid + 256]);
}

// ---------------- fp32 -> f16 transpose convert: dst[n*K+k] = src[k*N+n] ----------------
__global__ void cvt_f16_t(const float* __restrict__ s, _Float16* __restrict__ d,
                          int K, int N) {
  int i = blockIdx.x * 256 + threadIdx.x;   // linear index into dst
  if (i < K * N) {
    int n = i / K;
    int k = i - n * K;
    d[i] = (_Float16)s[(size_t)k * N + n];
  }
}

// ---------------- QKV GEMM: [4096,512] x Wt[1536,512] -> Q/K [B,H,T,64], V [B,H,64,T] f16 ----
__global__ void qkv_gemm(const _Float16* __restrict__ A, const _Float16* __restrict__ Wt,
                         _Float16* __restrict__ qb, _Float16* __restrict__ kb,
                         _Float16* __restrict__ vt) {
  __shared__ __attribute__((aligned(16))) _Float16 ash[4][16 * CDIM];  // 16KB per wave
  int lane  = threadIdx.x & 31;
  int wslot = threadIdx.x >> 5;
  int wid   = blockIdx.x * (blockDim.x >> 5) + (threadIdx.x >> 5);
  int mtile  = wid & 255;   // 256 M tiles of 16
  int nstrip = wid >> 8;    // 24 N strips of 64
  int hi = lane >> 4, lo = lane & 15;

  // Async-stage this wave's 16x512 A tile into LDS (32 x 512B chunks).
  unsigned ldsbase = (unsigned)(size_t)&ash[wslot][0];
  const _Float16* agbase = A + (size_t)(mtile * 16) * CDIM;
#pragma unroll
  for (int j = 0; j < 32; ++j)
    async_copy_b128(ldsbase + j * 512 + lane * 16, agbase + j * 256 + lane * 8);
  wait_async0();

  v8f c[4] = {};
  const _Float16* ap0 = &ash[wslot][lo * CDIM + hi * 8];
  const _Float16* wp0 = Wt + (size_t)(nstrip * 64 + lo) * CDIM + hi * 8;
  for (int k0 = 0; k0 < CDIM; k0 += 32) {
    __builtin_prefetch(wp0 + k0 + 64, 0, 3);
    v16h a = cat16(*(const v8h*)(ap0 + k0), *(const v8h*)(ap0 + k0 + 16));
#pragma unroll
    for (int sub = 0; sub < 4; ++sub) {
      const _Float16* wp = wp0 + (size_t)sub * 16 * CDIM + k0;
      v16h bf = cat16(*(const v8h*)wp, *(const v8h*)(wp + 16));
      c[sub] = wmma_f16(a, bf, c[sub]);
    }
  }
#pragma unroll
  for (int sub = 0; sub < 4; ++sub) {
    int col   = nstrip * 64 + sub * 16 + lo;
    int which = col >> 9;          // 0:q 1:k 2:v
    int inner = col & 511;
    int h = inner >> 6, d = inner & 63;
#pragma unroll
    for (int r = 0; r < 8; ++r) {
      int row = mtile * 16 + r + hi * 8;
      int bb  = row >> 11;         // /2048
      int t   = row & 2047;
      _Float16 val = (_Float16)c[sub][r];
      if (which == 0)
        qb[(((size_t)bb * NH + h) * TSEQ + t) * HD + d] = val;
      else if (which == 1)
        kb[(((size_t)bb * NH + h) * TSEQ + t) * HD + d] = val;
      else
        vt[(((size_t)bb * NH + h) * HD + d) * TSEQ + t] = val;   // V transposed [B,H,D,T]
    }
  }
}

// ---------------- Flash attention: one wave per (b,h,16-row q tile) ----------------
__global__ void attn_kernel(const _Float16* __restrict__ qg, const _Float16* __restrict__ kg,
                            const _Float16* __restrict__ vtg, _Float16* __restrict__ og) {
  __shared__ __attribute__((aligned(16))) _Float16 psh[4][16 * 32];
  int lane  = threadIdx.x & 31;
  int wslot = threadIdx.x >> 5;
  int wid   = blockIdx.x * 4 + wslot;     // 0..2047
  int qtile = wid & 127;
  int h     = (wid >> 7) & 7;
  int b     = wid >> 10;
  int hi = lane >> 4, lo = lane & 15;

  const _Float16* Q  = qg  + (((size_t)b * NH + h) * TSEQ + qtile * 16) * HD;
  const _Float16* K  = kg  + ((size_t)b * NH + h) * TSEQ * HD;
  const _Float16* Vt = vtg + ((size_t)b * NH + h) * HD * TSEQ;   // [D, T]

  // Q fragments: A layout, d = 0..31 and 32..63
  const _Float16* qp = Q + lo * HD + hi * 8;
  v16h qa0 = cat16(*(const v8h*)qp,        *(const v8h*)(qp + 16));
  v16h qa1 = cat16(*(const v8h*)(qp + 32), *(const v8h*)(qp + 48));

  float rowm[8], rowsum[8];
#pragma unroll
  for (int r = 0; r < 8; ++r) { rowm[r] = -3.0e38f; rowsum[r] = 0.0f; }
  v8f acc[4] = {};

  int kmax = ((qtile >> 4) + 1) * 256;     // block-causal: whole frames only
  for (int kc = 0; kc < kmax; kc += 32) {
    // S = Q K^T for 32 keys as two 16x16 tiles
    v8f s[2] = {};
#pragma unroll
    for (int sub = 0; sub < 2; ++sub) {
      const _Float16* kp = K + (size_t)(kc + sub * 16 + lo) * HD + hi * 8;
      v16h kb0 = cat16(*(const v8h*)kp,        *(const v8h*)(kp + 16));  // d 0..31
      v16h kb1 = cat16(*(const v8h*)(kp + 32), *(const v8h*)(kp + 48));  // d 32..63
      s[sub] = wmma_f16(qa0, kb0, s[sub]);
      s[sub] = wmma_f16(qa1, kb1, s[sub]);
    }
    // online softmax per row (row m = r + 8*hi lives in 16-lane half group)
#pragma unroll
    for (int r = 0; r < 8; ++r) {
      float e0 = s[0][r] * 0.125f;
      float e1 = s[1][r] * 0.125f;
      float mx = fmaxf(e0, e1);
      mx = fmaxf(mx, __shfl_xor(mx, 1, 32));
      mx = fmaxf(mx, __shfl_xor(mx, 2, 32));
      mx = fmaxf(mx, __shfl_xor(mx, 4, 32));
      mx = fmaxf(mx, __shfl_xor(mx, 8, 32));
      float mnew = fmaxf(rowm[r], mx);
      float fr   = __expf(rowm[r] - mnew);
      rowm[r] = mnew;
      float p0 = __expf(e0 - mnew);
      float p1 = __expf(e1 - mnew);
      float ls = p0 + p1;
      ls += __shfl_xor(ls, 1, 32);
      ls += __shfl_xor(ls, 2, 32);
      ls += __shfl_xor(ls, 4, 32);
      ls += __shfl_xor(ls, 8, 32);
      rowsum[r] = rowsum[r] * fr + ls;
#pragma unroll
      for (int d = 0; d < 4; ++d) acc[d][r] = acc[d][r] * fr;
      int m = r + hi * 8;
      psh[wslot][m * 32 + lo]      = (_Float16)p0;
      psh[wslot][m * 32 + 16 + lo] = (_Float16)p1;
    }
    asm volatile("s_wait_dscnt 0x0" ::: "memory");
    // reload P as an A fragment (16x32) from LDS
    const _Float16* pp = &psh[wslot][lo * 32 + hi * 8];
    v16h pa = cat16(*(const v8h*)pp, *(const v8h*)(pp + 16));
    // O += P * V : Vt is [D, T], so per d-column keys are contiguous
#pragma unroll
    for (int d = 0; d < 4; ++d) {
      const _Float16* vp = Vt + (size_t)(d * 16 + lo) * TSEQ + kc + hi * 8;
      v16h vbf = cat16(*(const v8h*)vp, *(const v8h*)(vp + 16));
      acc[d] = wmma_f16(pa, vbf, acc[d]);
    }
  }
  // finalize: divide by rowsum, write [b, t, h*64+d] f16
#pragma unroll
  for (int r = 0; r < 8; ++r) {
    float inv = 1.0f / rowsum[r];
    int m = r + hi * 8;
    int t = qtile * 16 + m;
#pragma unroll
    for (int d = 0; d < 4; ++d) {
      og[((size_t)b * TSEQ + t) * INNER + h * 64 + d * 16 + lo] =
          (_Float16)(acc[d][r] * inv);
    }
  }
}

// ---------------- Output GEMM: [4096,512] x Wt[512,512] + bias -> fp32 ----------------
__global__ void out_gemm(const _Float16* __restrict__ A, const _Float16* __restrict__ Wt,
                         const float* __restrict__ bias, float* __restrict__ out) {
  __shared__ __attribute__((aligned(16))) _Float16 ash[4][16 * INNER];  // 16KB per wave
  int lane  = threadIdx.x & 31;
  int wslot = threadIdx.x >> 5;
  int wid   = blockIdx.x * (blockDim.x >> 5) + (threadIdx.x >> 5);
  int mtile  = wid >> 3;   // 256 M tiles
  int nstrip = wid & 7;    // 8 N strips of 64
  int hi = lane >> 4, lo = lane & 15;

  // Async-stage this wave's 16x512 A tile into LDS.
  unsigned ldsbase = (unsigned)(size_t)&ash[wslot][0];
  const _Float16* agbase = A + (size_t)(mtile * 16) * INNER;
#pragma unroll
  for (int j = 0; j < 32; ++j)
    async_copy_b128(ldsbase + j * 512 + lane * 16, agbase + j * 256 + lane * 8);
  wait_async0();

  v8f c[4] = {};
  const _Float16* ap0 = &ash[wslot][lo * INNER + hi * 8];
  const _Float16* wp0 = Wt + (size_t)(nstrip * 64 + lo) * INNER + hi * 8;
  for (int k0 = 0; k0 < INNER; k0 += 32) {
    __builtin_prefetch(wp0 + k0 + 64, 0, 3);
    v16h a = cat16(*(const v8h*)(ap0 + k0), *(const v8h*)(ap0 + k0 + 16));
#pragma unroll
    for (int sub = 0; sub < 4; ++sub) {
      const _Float16* wp = wp0 + (size_t)sub * 16 * INNER + k0;
      v16h bf = cat16(*(const v8h*)wp, *(const v8h*)(wp + 16));
      c[sub] = wmma_f16(a, bf, c[sub]);
    }
  }
#pragma unroll
  for (int sub = 0; sub < 4; ++sub) {
    int col = nstrip * 64 + sub * 16 + lo;
    float bv = bias[col];
#pragma unroll
    for (int r = 0; r < 8; ++r) {
      int row = mtile * 16 + r + hi * 8;
      out[(size_t)row * CDIM + col] = c[sub][r] + bv;
    }
  }
}

extern "C" void kernel_launch(void* const* d_in, const int* in_sizes, int n_in,
                              void* d_out, int out_size, void* d_ws, size_t ws_size,
                              hipStream_t stream) {
  (void)in_sizes; (void)n_in; (void)out_size; (void)ws_size;
  const float* x    = (const float*)d_in[0];
  const float* g    = (const float*)d_in[1];
  const float* be   = (const float*)d_in[2];
  const float* wqkv = (const float*)d_in[3];
  const float* wout = (const float*)d_in[4];
  const float* bout = (const float*)d_in[5];
  float* out = (float*)d_out;

  char* wsp = (char*)d_ws;
  size_t off = 0;
  auto alloc = [&](size_t bytes) -> void* {
    void* p = (void*)(wsp + off);
    off += (bytes + 255) & ~(size_t)255;
    return p;
  };
  _Float16* xn16 = (_Float16*)alloc((size_t)BATCH * TSEQ * CDIM * 2);
  _Float16* wq16 = (_Float16*)alloc((size_t)CDIM * NQKV * 2);   // transposed [1536,512]
  _Float16* wo16 = (_Float16*)alloc((size_t)INNER * CDIM * 2);  // transposed [512,512]
  _Float16* qb   = (_Float16*)alloc((size_t)BATCH * NH * TSEQ * HD * 2);
  _Float16* kb   = (_Float16*)alloc((size_t)BATCH * NH * TSEQ * HD * 2);
  _Float16* vt   = (_Float16*)alloc((size_t)BATCH * NH * HD * TSEQ * 2); // [B,H,D,T]
  _Float16* ao16 = (_Float16*)alloc((size_t)BATCH * TSEQ * INNER * 2);

  ln_to_f16<<<BATCH * TSEQ, 256, 0, stream>>>(x, g, be, xn16);
  cvt_f16_t<<<(CDIM * NQKV + 255) / 256, 256, 0, stream>>>(wqkv, wq16, CDIM, NQKV);
  cvt_f16_t<<<(INNER * CDIM + 255) / 256, 256, 0, stream>>>(wout, wo16, INNER, CDIM);
  // 256*24 = 6144 waves, 4 waves/block
  qkv_gemm<<<1536, 128, 0, stream>>>(xn16, wq16, qb, kb, vt);
  // 2*8*128 = 2048 waves, 4 waves/block
  attn_kernel<<<512, 128, 0, stream>>>(qb, kb, vt, ao16);
  // 256*8 = 2048 waves, 4 waves/block
  out_gemm<<<512, 128, 0, stream>>>(ao16, wo16, bout, out);
}
